// Decoder_62483184222320
// MI455X (gfx1250) — matrix operand order repaired and checked by reading, not compiled, not run on hardware
//
#include <hip/hip_runtime.h>

typedef __attribute__((ext_vector_type(16))) _Float16 v16h;
typedef __attribute__((ext_vector_type(8)))  float    v8f;

#define NB      8
#define NC      128
#define IMG     64
#define HW      4096          // 64*64
#define NIN     384           // 3*NC
#define NCALLS  32
#define LAT     512
#define WCOLS   49152         // 3*NC*NC
#define NPARAMS 49280         // WCOLS + NC
#define CLIP_SZ (NB*3*HW)
#define EMB_SLICE (NB*NC*HW)  // 4194304 floats per step slice

// ---------------------------------------------------------------------------
// Hyper-network: params = lat @ w_dyn + b_dyn, emitted directly as
// f16 weights w16[b][o][i] (o=col/384, i=col%384 => linear index b*WCOLS+col)
// and f32 bias[b][o].
// ---------------------------------------------------------------------------
__global__ void __launch_bounds__(256)
k_params(const float* __restrict__ lat, const float* __restrict__ w_dyn,
         const float* __restrict__ b_dyn, _Float16* __restrict__ w16,
         float* __restrict__ bias) {
    __shared__ float slat[NB * LAT];
    for (int j = threadIdx.x; j < NB * LAT; j += 256) slat[j] = lat[j];
    __syncthreads();
    int col = blockIdx.x * 256 + threadIdx.x;
    if (col >= NPARAMS) return;
    float acc[NB] = {};
    for (int k = 0; k < LAT; ++k) {
        float wv = w_dyn[(size_t)k * NPARAMS + col];   // coalesced over col
        #pragma unroll
        for (int b = 0; b < NB; ++b) acc[b] += slat[b * LAT + k] * wv;
    }
    float bd = b_dyn[col];
    if (col < WCOLS) {
        #pragma unroll
        for (int b = 0; b < NB; ++b)
            w16[(size_t)b * WCOLS + col] = (_Float16)(acc[b] + bd);
    } else {
        int o = col - WCOLS;
        #pragma unroll
        for (int b = 0; b < NB; ++b) bias[b * NC + o] = acc[b] + bd;
    }
}

// ---------------------------------------------------------------------------
// Seed state -> embs[0]: zeros with center pixel = 1 for every (b, c).
// ---------------------------------------------------------------------------
__global__ void __launch_bounds__(256)
k_seed(float* __restrict__ embs0) {
    int idx = blockIdx.x * 256 + threadIdx.x;
    if (idx >= EMB_SLICE) return;
    int pix = idx & (HW - 1);
    embs0[idx] = (pix == (32 * IMG + 32)) ? 1.0f : 0.0f;
}

// ---------------------------------------------------------------------------
// Sobel perception + instance-norm statistics.
// One block per (b, c) plane: compute gx, gy (zero-padded SAME conv, XLA
// cross-correlation convention), store as f16, and reduce mean/rstd for
// x, gx, gy into stats[b][{c, 128+c, 256+c}][2].
// ---------------------------------------------------------------------------
__global__ void __launch_bounds__(256)
k_sobel_stats(const float* __restrict__ x_base, _Float16* __restrict__ gx,
              _Float16* __restrict__ gy, float* __restrict__ stats) {
    int bc = blockIdx.x;                         // b*128 + c
    const float* x = x_base + (size_t)bc * HW;
    _Float16* gxo = gx + (size_t)bc * HW;
    _Float16* gyo = gy + (size_t)bc * HW;

    float s0 = 0.f, s1 = 0.f, s2 = 0.f, s3 = 0.f, s4 = 0.f, s5 = 0.f;
    for (int k = 0; k < HW / 256; ++k) {
        int p = threadIdx.x + k * 256;
        int r = p >> 6, c = p & 63;
        float x00 = (r > 0  && c > 0 ) ? x[p - 65] : 0.f;
        float x01 = (r > 0           ) ? x[p - 64] : 0.f;
        float x02 = (r > 0  && c < 63) ? x[p - 63] : 0.f;
        float x10 = (c > 0           ) ? x[p - 1 ] : 0.f;
        float x12 = (c < 63          ) ? x[p + 1 ] : 0.f;
        float x20 = (r < 63 && c > 0 ) ? x[p + 63] : 0.f;
        float x21 = (r < 63          ) ? x[p + 64] : 0.f;
        float x22 = (r < 63 && c < 63) ? x[p + 65] : 0.f;
        float xc  = x[p];
        float gxv = (x02 - x00 + 2.f * (x12 - x10) + x22 - x20) * 0.125f;
        float gyv = (x20 - x00 + 2.f * (x21 - x01) + x22 - x02) * 0.125f;
        gxo[p] = (_Float16)gxv;
        gyo[p] = (_Float16)gyv;
        s0 += xc;  s1 += xc  * xc;
        s2 += gxv; s3 += gxv * gxv;
        s4 += gyv; s5 += gyv * gyv;
    }
    float v[6] = {s0, s1, s2, s3, s4, s5};
    #pragma unroll
    for (int i = 0; i < 6; ++i)
        #pragma unroll
        for (int m = 16; m >= 1; m >>= 1) v[i] += __shfl_xor(v[i], m, 32);

    __shared__ float red[8][6];
    int wave = threadIdx.x >> 5, lane = threadIdx.x & 31;
    if (lane == 0)
        for (int i = 0; i < 6; ++i) red[wave][i] = v[i];
    __syncthreads();
    if (threadIdx.x == 0) {
        float t[6] = {};
        for (int w = 0; w < 8; ++w)
            for (int i = 0; i < 6; ++i) t[i] += red[w][i];
        int b = bc >> 7, c = bc & 127;
        const float inv = 1.0f / (float)HW;
        #pragma unroll
        for (int g = 0; g < 3; ++g) {
            float m   = t[2 * g] * inv;
            float var = t[2 * g + 1] * inv - m * m;
            int i = b * NIN + g * NC + c;
            stats[2 * i]     = m;
            stats[2 * i + 1] = rsqrtf(var + 1e-5f);
        }
    }
}

// ---------------------------------------------------------------------------
// Fused normalize + einsum (WMMA f16) + leaky residual.
// One block = one batch x 64-pixel strip. 8 waves, wave w owns output
// channels [16w, 16w+16). K = 384 -> 12 x v_wmma_f32_16x16x32_f16 per tile.
// Reads state from embs[s], writes embs[s+1] (state lives in d_out / L2).
// ---------------------------------------------------------------------------
__global__ void __launch_bounds__(256)
k_gemm_step(const float* __restrict__ prev, float* __restrict__ next,
            const _Float16* __restrict__ gx, const _Float16* __restrict__ gy,
            const float* __restrict__ stats, const _Float16* __restrict__ w16,
            const float* __restrict__ bias, const float* __restrict__ lfp) {
    __shared__ _Float16 yn[64][NIN + 8];   // [pixel][input-channel], padded

    int b  = blockIdx.x >> 6;
    int pb = blockIdx.x & 63;
    int p0 = pb * 64;

    // Stage normalized perception tile: yn[p][i] = (src - mean_i) * rstd_i
    for (int j = threadIdx.x; j < 64 * NIN; j += 256) {
        int i = j >> 6, p = j & 63;          // consecutive threads -> pixels
        int pix = p0 + p;
        float val;
        if (i < NC)          val = prev[((size_t)(b * NC + i)) * HW + pix];
        else if (i < 2 * NC) val = (float)gx[((size_t)(b * NC + i - NC)) * HW + pix];
        else                 val = (float)gy[((size_t)(b * NC + i - 2 * NC)) * HW + pix];
        float m = stats[2 * (b * NIN + i)];
        float r = stats[2 * (b * NIN + i) + 1];
        yn[p][i] = (_Float16)((val - m) * r);
    }
    __syncthreads();

    int wave = threadIdx.x >> 5, lane = threadIdx.x & 31;
    int lg = (lane >= 16) ? 1 : 0;          // lane group
    int ln = lane & 15;
    int o0 = wave * 16 + ln;                // A-matrix row held by this lane

    // A fragments: 16-bit A 16x32 layout = two contiguous 8-half runs per lane
    // at K-offsets {kb*32 + lg*8, kb*32 + lg*8 + 16}. Kept in VGPRs (12x32B).
    union AF { v16h v; uint4 u[2]; };
    AF a[12];
    const uint4* wp = (const uint4*)(w16 + (size_t)(b * NC + o0) * NIN);
    #pragma unroll
    for (int kb = 0; kb < 12; ++kb) {
        a[kb].u[0] = wp[kb * 4 + lg];        // halves K = kb*32+lg*8 .. +7
        a[kb].u[1] = wp[kb * 4 + lg + 2];    // halves K = kb*32+lg*8+16 .. +7
    }

    float lf = fminf(fmaxf(lfp[0], 0.001f), 1000.0f);

    for (int nt = 0; nt < 4; ++nt) {
        v8f acc = {};
        int pN = nt * 16 + ln;               // pixel column for B / D
        #pragma unroll
        for (int kb = 0; kb < 12; ++kb) {
            // B 32x16 layout: same K order as A, two 8-half runs per lane at
            // K-offsets {lg*8, lg*8 + 16} -> two ds_load_b128.
            union BF { v16h v; uint4 u[2]; } bf;
            const uint4* bp = (const uint4*)(&yn[pN][kb * 32 + lg * 8]);
            bf.u[0] = bp[0];
            bf.u[1] = bp[2];
            acc = __builtin_amdgcn_wmma_f32_16x16x32_f16(
                false, a[kb].v, false, bf.v, (short)0, acc, false, false);
        }
        // Epilogue: out = prev + lf * (Y + bias); D layout row = r + lg*8.
        int pix = p0 + nt * 16 + ln;
        #pragma unroll
        for (int r8 = 0; r8 < 8; ++r8) {
            int o = wave * 16 + r8 + lg * 8;
            size_t idx = ((size_t)(b * NC + o)) * HW + pix;
            next[idx] = prev[idx] + lf * (acc[r8] + bias[b * NC + o]);
        }
    }
}

// ---------------------------------------------------------------------------
// Final outputs: clipped RGB + raw RGB from embs[32].
// ---------------------------------------------------------------------------
__global__ void __launch_bounds__(256)
k_final(const float* __restrict__ last, float* __restrict__ clipd,
        float* __restrict__ raw) {
    int idx = blockIdx.x * 256 + threadIdx.x;
    if (idx >= CLIP_SZ) return;
    int b = idx / (3 * HW);
    int rest = idx - b * 3 * HW;             // c*HW + pix, c < 3
    float v = last[(size_t)b * NC * HW + rest];
    clipd[idx] = fminf(fmaxf(v, -1.f), 1.f);
    raw[idx] = v;
}

extern "C" void kernel_launch(void* const* d_in, const int* in_sizes, int n_in,
                              void* d_out, int out_size, void* d_ws, size_t ws_size,
                              hipStream_t stream) {
    const float* lat   = (const float*)d_in[0];
    const float* lfp   = (const float*)d_in[1];
    const float* w_dyn = (const float*)d_in[2];
    const float* b_dyn = (const float*)d_in[3];

    float* out   = (float*)d_out;
    float* clipd = out;
    float* embs  = out + CLIP_SZ;                       // 33 slices of EMB_SLICE
    float* raw   = out + CLIP_SZ + (size_t)(NCALLS + 1) * EMB_SLICE;

    // Workspace carve-out (~17.6 MB): gx|gy (f16 planes), stats, w16, bias.
    char* ws = (char*)d_ws;
    _Float16* gx   = (_Float16*)ws;
    _Float16* gy   = gx + (size_t)NB * NC * HW;
    float*    stats = (float*)(gy + (size_t)NB * NC * HW);
    _Float16* w16  = (_Float16*)(stats + NB * NIN * 2);
    float*    bias = (float*)(w16 + (size_t)NB * WCOLS);

    hipLaunchKernelGGL(k_params, dim3((NPARAMS + 255) / 256), dim3(256), 0, stream,
                       lat, w_dyn, b_dyn, w16, bias);
    hipLaunchKernelGGL(k_seed, dim3(EMB_SLICE / 256), dim3(256), 0, stream, embs);

    for (int s = 0; s < NCALLS; ++s) {
        float* prev = embs + (size_t)s * EMB_SLICE;
        float* next = prev + EMB_SLICE;
        hipLaunchKernelGGL(k_sobel_stats, dim3(NB * NC), dim3(256), 0, stream,
                           prev, gx, gy, stats);
        hipLaunchKernelGGL(k_gemm_step, dim3(NB * 64), dim3(256), 0, stream,
                           prev, next, gx, gy, stats, w16, bias, lfp);
    }

    hipLaunchKernelGGL(k_final, dim3((CLIP_SZ + 255) / 256), dim3(256), 0, stream,
                       embs + (size_t)NCALLS * EMB_SLICE, clipd, raw);
}
